// SplineLayer_89026082111590
// MI455X (gfx1250) — matrix operand order repaired and checked by reading, not compiled, not run on hardware
//
#include <hip/hip_runtime.h>
#include <hip/hip_bf16.h>

// ---------------------------------------------------------------------------
// SplineConv GNN layer for MI455X (gfx1250), wave32 + WMMA bf16.
//   scatter (f32 global atomics)  ->  GEMM via v_wmma_f32_16x16x32_bf16
//   BN stats via LDS tree + atomics ->  fused scale/shift epilogues
// ---------------------------------------------------------------------------

#define NKIN   30
#define C1IN   32
#define OUTC   64
#define KT     25
#define BN_EPS 1e-5f

typedef __attribute__((ext_vector_type(16))) __bf16 v16bf;
typedef __attribute__((ext_vector_type(8)))  float  v8f;
typedef __attribute__((ext_vector_type(8)))  float  f32x8;

// ---------------- elementwise / prep kernels -------------------------------

__global__ void k_xin(const float* __restrict__ x, const float* __restrict__ pos,
                      float* __restrict__ xin, int n) {
  int i = blockIdx.x * blockDim.x + threadIdx.x;
  if (i >= n * C1IN) return;
  int row = i >> 5, c = i & 31;
  xin[i] = (c < NKIN) ? x[row * NKIN + c] : pos[row * 2 + (c - NKIN)];
}

__global__ void k_deg(const int* __restrict__ dstI, float* __restrict__ deg, int e) {
  int i = blockIdx.x * blockDim.x + threadIdx.x;
  if (i < e) atomicAdd(&deg[dstI[i]], 1.0f);
}

__global__ void k_rdeg(float* __restrict__ deg, int n) {
  int i = blockIdx.x * blockDim.x + threadIdx.x;
  if (i < n) deg[i] = 1.0f / fmaxf(deg[i], 1.0f);
}

// src f32 [KA][64]  ->  dst bf16 [64][KA]   (transpose so B-fragments are contiguous)
__global__ void k_convert_w(const float* __restrict__ src, __bf16* __restrict__ dst, int ka) {
  int i = blockIdx.x * blockDim.x + threadIdx.x;
  if (i >= ka * OUTC) return;
  int k = i >> 6, c = i & 63;
  dst[(size_t)c * ka + k] = (__bf16)src[i];
}

// ---------------- scatter-accumulate (one wave per edge) -------------------

template<int CIN>
__global__ void k_scatter(const float* __restrict__ xsrc, const int* __restrict__ srcI,
                          const int* __restrict__ dstI, const float* __restrict__ attr,
                          float* __restrict__ acc, int e) {
  int wave = (int)((blockIdx.x * blockDim.x + threadIdx.x) >> 5);
  if (wave >= e) return;
  int lane = threadIdx.x & 31;
  int s = srcI[wave], d = dstI[wave];
  float vx = attr[2 * wave]     * 4.0f;
  float vy = attr[2 * wave + 1] * 4.0f;
  float k0x = fminf(fmaxf(floorf(vx), 0.0f), 3.0f);
  float k0y = fminf(fmaxf(floorf(vy), 0.0f), 3.0f);
  float fx = vx - k0x, fy = vy - k0y;
  int ix = (int)k0x, iy = (int)k0y;
  float bs[4] = { (1.0f - fx) * (1.0f - fy), fx * (1.0f - fy),
                  (1.0f - fx) * fy,          fx * fy };
  int   wi[4] = { ix + 5 * iy, ix + 1 + 5 * iy,
                  ix + 5 * (iy + 1), ix + 1 + 5 * (iy + 1) };
  float x0 = xsrc[(size_t)s * CIN + lane];
  float x1 = (CIN == 64) ? xsrc[(size_t)s * CIN + lane + 32] : 0.0f;
#pragma unroll
  for (int t = 0; t < 4; ++t) {
    float* p = acc + ((size_t)d * KT + wi[t]) * CIN;
    atomicAdd(p + lane, bs[t] * x0);
    if (CIN == 64) atomicAdd(p + lane + 32, bs[t] * x1);
  }
}

// ---------------- GEMM: out[N,64] = (A @ Wb) * rdeg + R @ Rb ---------------
// One wave computes a 16-row x 64-col strip (4 WMMA column tiles, A reused in regs).

__device__ __forceinline__ v16bf load_a_bf16(const float* arow, int k0, int abase) {
  f32x8 lo = *(const f32x8*)(arow + k0 + abase);
  f32x8 hi = *(const f32x8*)(arow + k0 + 8 + abase);
  v16bf a;
#pragma unroll
  for (int j = 0; j < 8; ++j) { a[j] = (__bf16)lo[j]; a[8 + j] = (__bf16)hi[j]; }
  return a;
}

__global__ void k_gemm_bf16(const float* __restrict__ A, const __bf16* __restrict__ Wb, int KA,
                            const float* __restrict__ R, const __bf16* __restrict__ Rb, int KR,
                            const float* __restrict__ rdeg, float* __restrict__ out, int mtiles) {
  int wave = threadIdx.x >> 5;
  int lane = threadIdx.x & 31;
  int tile = blockIdx.x * (blockDim.x >> 5) + wave;   // wave-uniform
  if (tile >= mtiles) return;                          // EXEC stays all-ones for WMMA
  int m0    = tile * 16;
  int lm    = lane & 15;
  int half  = (lane < 16) ? 0 : 1;
  int abase = half * 16;
  int bbase = half * 16;
  int row   = m0 + lm;

  v8f c[4] = {v8f{}, v8f{}, v8f{}, v8f{}};

  if (A) {
    const float* arow = A + (size_t)row * KA;
    for (int k0 = 0; k0 < KA; k0 += 32) {
      if (k0 + 32 < KA) __builtin_prefetch(arow + k0 + 32, 0, 1);   // global_prefetch_b8
      v16bf a = load_a_bf16(arow, k0, abase);
#pragma unroll
      for (int t = 0; t < 4; ++t) {
        v16bf b = *(const v16bf*)(Wb + (size_t)(t * 16 + lm) * KA + k0 + bbase);
        c[t] = __builtin_amdgcn_wmma_f32_16x16x32_bf16(false, a, false, b,
                                                       (short)0, c[t], false, false);
      }
    }
  }

  if (rdeg) {                     // mean aggregation: scale acc part only
#pragma unroll
    for (int r = 0; r < 8; ++r) {
      float s = rdeg[m0 + r + half * 8];
      c[0][r] *= s; c[1][r] *= s; c[2][r] *= s; c[3][r] *= s;
    }
  }

  if (R) {                        // root-weight term
    const float* rrow = R + (size_t)row * KR;
    for (int k0 = 0; k0 < KR; k0 += 32) {
      v16bf a = load_a_bf16(rrow, k0, abase);
#pragma unroll
      for (int t = 0; t < 4; ++t) {
        v16bf b = *(const v16bf*)(Rb + (size_t)(t * 16 + lm) * KR + k0 + bbase);
        c[t] = __builtin_amdgcn_wmma_f32_16x16x32_bf16(false, a, false, b,
                                                       (short)0, c[t], false, false);
      }
    }
  }

#pragma unroll
  for (int r = 0; r < 8; ++r) {
    int orow = m0 + r + half * 8;
    float* op = out + (size_t)orow * OUTC + lm;
    op[0] = c[0][r]; op[16] = c[1][r]; op[32] = c[2][r]; op[48] = c[3][r];
  }
}

// ---------------- batch-norm stats + epilogues -----------------------------

__global__ void k_stats(const float* __restrict__ h, int n,
                        float* __restrict__ sum, float* __restrict__ sumsq) {
  __shared__ float rs[256], rq[256];
  int t = threadIdx.x;
  int c = t & 63, g = t >> 6;
  float s = 0.0f, q = 0.0f;
  for (int row = blockIdx.x * 4 + g; row < n; row += gridDim.x * 4) {
    float v = h[(size_t)row * OUTC + c];
    s += v; q += v * v;
  }
  rs[t] = s; rq[t] = q; __syncthreads();
  if (t < 128) { rs[t] += rs[t + 128]; rq[t] += rq[t + 128]; } __syncthreads();
  if (t < 64) {
    atomicAdd(&sum[c],   rs[t] + rs[t + 64]);
    atomicAdd(&sumsq[c], rq[t] + rq[t + 64]);
  }
}

__global__ void k_bnfin(const float* __restrict__ sum, const float* __restrict__ sumsq,
                        const float* __restrict__ g, const float* __restrict__ b, int n,
                        float* __restrict__ scale, float* __restrict__ shift) {
  int c = threadIdx.x;
  if (c >= OUTC) return;
  float inv = 1.0f / (float)n;
  float mu  = sum[c] * inv;
  float var = fmaxf(sumsq[c] * inv - mu * mu, 0.0f);
  float sc  = g[c] * rsqrtf(var + BN_EPS);
  scale[c] = sc;
  shift[c] = b[c] - mu * sc;
}

__global__ void k_apply_relu(const float* __restrict__ pre, const float* __restrict__ scale,
                             const float* __restrict__ shift, float* __restrict__ out, int total) {
  int i = blockIdx.x * blockDim.x + threadIdx.x;
  if (i >= total) return;
  int c = i & 63;
  out[i] = fmaxf(pre[i] * scale[c] + shift[c], 0.0f);
}

__global__ void k_out(const float* __restrict__ pre2, const float* __restrict__ sk,
                      const float* __restrict__ s2, const float* __restrict__ h2,
                      const float* __restrict__ s3, const float* __restrict__ h3,
                      float* __restrict__ out, int total) {
  int i = blockIdx.x * blockDim.x + threadIdx.x;
  if (i >= total) return;
  int c = i & 63;
  float v = pre2[i] * s2[c] + h2[c] + sk[i] * s3[c] + h3[c];
  out[i] = fmaxf(v, 0.0f);
}

// ---------------- host-side orchestration ----------------------------------

extern "C" void kernel_launch(void* const* d_in, const int* in_sizes, int n_in,
                              void* d_out, int out_size, void* d_ws, size_t ws_size,
                              hipStream_t stream) {
  const float* x     = (const float*)d_in[0];
  const float* pos   = (const float*)d_in[1];
  const float* attr  = (const float*)d_in[2];
  const int*   eidx  = (const int*)  d_in[3];
  const float* W1    = (const float*)d_in[4];
  const float* root1 = (const float*)d_in[5];
  const float* g1    = (const float*)d_in[6];
  const float* b1    = (const float*)d_in[7];
  const float* W2    = (const float*)d_in[8];
  const float* root2 = (const float*)d_in[9];
  const float* g2    = (const float*)d_in[10];
  const float* b2    = (const float*)d_in[11];
  const float* Wskip = (const float*)d_in[12];
  const float* g3    = (const float*)d_in[13];
  const float* b3    = (const float*)d_in[14];
  float* outp = (float*)d_out;

  const int N = in_sizes[0] / NKIN;
  const int E = in_sizes[3] / 2;
  const int* srcI = eidx;
  const int* dstI = eidx + E;
  const int KA1 = KT * C1IN;   // 800
  const int KA2 = KT * OUTC;   // 1600

  // ---- workspace carve (256B aligned bump allocator) ----
  char* wp = (char*)d_ws;
  auto carve = [&](size_t bytes) -> char* {
    char* r = wp; wp += (bytes + 255) & ~(size_t)255; return r;
  };
  float*  acc     = (float*) carve((size_t)N * KT * OUTC * 4);  // 320 MB, reused both layers
  float*  xin     = (float*) carve((size_t)N * C1IN * 4);
  float*  h1      = (float*) carve((size_t)N * OUTC * 4);
  float*  pre1    = (float*) carve((size_t)N * OUTC * 4);
  float*  pre2    = (float*) carve((size_t)N * OUTC * 4);
  float*  skipPre = (float*) carve((size_t)N * OUTC * 4);
  float*  rdeg    = (float*) carve((size_t)N * 4);
  __bf16* Wb1     = (__bf16*)carve((size_t)OUTC * KA1 * 2);
  __bf16* Wb2     = (__bf16*)carve((size_t)OUTC * KA2 * 2);
  __bf16* rootb1  = (__bf16*)carve((size_t)OUTC * C1IN * 2);
  __bf16* rootb2  = (__bf16*)carve((size_t)OUTC * OUTC * 2);
  __bf16* wskipb  = (__bf16*)carve((size_t)OUTC * C1IN * 2);
  float*  stats   = (float*) carve((size_t)3 * 4 * OUTC * 4);   // per stage: sum,sumsq,scale,shift
  float* sum1 = stats + 0 * 256, *sq1 = sum1 + 64, *sc1 = sum1 + 128, *sh1 = sum1 + 192;
  float* sum2 = stats + 1 * 256, *sq2 = sum2 + 64, *sc2 = sum2 + 128, *sh2 = sum2 + 192;
  float* sum3 = stats + 2 * 256, *sq3 = sum3 + 64, *sc3 = sum3 + 128, *sh3 = sum3 + 192;

  const int BT = 256;
  const int mtiles = N / 16;                          // 3125 (N divisible by 16)
  const int gemmBlocks = (mtiles + 3) / 4;            // 4 waves / block
  auto cdiv = [](int a, int b) { return (a + b - 1) / b; };

  hipMemsetAsync(rdeg,  0, (size_t)N * 4, stream);
  hipMemsetAsync(stats, 0, (size_t)3 * 4 * OUTC * 4, stream);

  // prep
  k_xin <<<cdiv(N * C1IN, BT), BT, 0, stream>>>(x, pos, xin, N);
  k_deg <<<cdiv(E, BT), BT, 0, stream>>>(dstI, rdeg, E);
  k_rdeg<<<cdiv(N, BT), BT, 0, stream>>>(rdeg, N);
  k_convert_w<<<cdiv(KA1 * OUTC, BT), BT, 0, stream>>>(W1,    Wb1,    KA1);
  k_convert_w<<<cdiv(KA2 * OUTC, BT), BT, 0, stream>>>(W2,    Wb2,    KA2);
  k_convert_w<<<cdiv(C1IN * OUTC, BT), BT, 0, stream>>>(root1, rootb1, C1IN);
  k_convert_w<<<cdiv(OUTC * OUTC, BT), BT, 0, stream>>>(root2, rootb2, OUTC);
  k_convert_w<<<cdiv(C1IN * OUTC, BT), BT, 0, stream>>>(Wskip, wskipb, C1IN);

  // skip path: skipPre = xin @ Wskip ; BN stats
  k_gemm_bf16<<<gemmBlocks, 128, 0, stream>>>(nullptr, nullptr, 0,
                                              xin, wskipb, C1IN, nullptr, skipPre, mtiles);
  k_stats<<<256, 256, 0, stream>>>(skipPre, N, sum3, sq3);
  k_bnfin<<<1, 64, 0, stream>>>(sum3, sq3, g3, b3, N, sc3, sh3);

  // layer 1
  hipMemsetAsync(acc, 0, (size_t)N * KT * C1IN * 4, stream);
  k_scatter<C1IN><<<cdiv(E, 8), BT, 0, stream>>>(xin, srcI, dstI, attr, acc, E);
  k_gemm_bf16<<<gemmBlocks, 128, 0, stream>>>(acc, Wb1, KA1,
                                              xin, rootb1, C1IN, rdeg, pre1, mtiles);
  k_stats<<<256, 256, 0, stream>>>(pre1, N, sum1, sq1);
  k_bnfin<<<1, 64, 0, stream>>>(sum1, sq1, g1, b1, N, sc1, sh1);
  k_apply_relu<<<cdiv(N * OUTC, BT), BT, 0, stream>>>(pre1, sc1, sh1, h1, N * OUTC);

  // layer 2
  hipMemsetAsync(acc, 0, (size_t)N * KT * OUTC * 4, stream);
  k_scatter<OUTC><<<cdiv(E, 8), BT, 0, stream>>>(h1, srcI, dstI, attr, acc, E);
  k_gemm_bf16<<<gemmBlocks, 128, 0, stream>>>(acc, Wb2, KA2,
                                              h1, rootb2, OUTC, rdeg, pre2, mtiles);
  k_stats<<<256, 256, 0, stream>>>(pre2, N, sum2, sq2);
  k_bnfin<<<1, 64, 0, stream>>>(sum2, sq2, g2, b2, N, sc2, sh2);

  // fused output: relu(bn2(pre2) + bn3(skipPre))
  k_out<<<cdiv(N * OUTC, BT), BT, 0, stream>>>(pre2, skipPre, sc2, sh2, sc3, sh3,
                                               outp, N * OUTC);
}
